// StableResonatorBank_68195490726314
// MI455X (gfx1250) — compile-verified, matrix operand order checked by reading, not run on hardware
//
#include <hip/hip_runtime.h>
#include <math.h>

// Problem constants (B,S,D,R) from the reference setup_inputs().
#define B_ 8
#define S_ 4096
#define D_ 1024
#define R_ 256

typedef __attribute__((ext_vector_type(16))) _Float16 v16h;
typedef __attribute__((ext_vector_type(8)))  _Float16 v8h;
typedef __attribute__((ext_vector_type(4)))  _Float16 v4h;
typedef __attribute__((ext_vector_type(8)))  float    v8f;

// ---- WMMA fragment loaders (CDNA5 §7.12.2 16-bit A/B layout) -------------
// Lane (l&15) owns one row/column; halves [0..7] = K[k0..k0+7],
// halves [8..15] = K[k0+16..k0+23], with k0 already including 8*(l>>4).
__device__ __forceinline__ v16h load_frag_h(const _Float16* __restrict__ p) {
  v8h lo = *(const v8h*)(p);
  v8h hi = *(const v8h*)(p + 16);
  v16h r;
#pragma unroll
  for (int i = 0; i < 8; ++i) { r[i] = lo[i]; r[i + 8] = hi[i]; }
  return r;
}

// Same, but from f32 memory with an on-the-fly scale (fused L2 normalize).
__device__ __forceinline__ v16h load_frag_f32_scaled(const float* __restrict__ p, float s) {
  float4 a0 = *(const float4*)(p);
  float4 a1 = *(const float4*)(p + 4);
  float4 b0 = *(const float4*)(p + 16);
  float4 b1 = *(const float4*)(p + 20);
  v16h r;
  r[0]  = (_Float16)(a0.x * s); r[1]  = (_Float16)(a0.y * s);
  r[2]  = (_Float16)(a0.z * s); r[3]  = (_Float16)(a0.w * s);
  r[4]  = (_Float16)(a1.x * s); r[5]  = (_Float16)(a1.y * s);
  r[6]  = (_Float16)(a1.z * s); r[7]  = (_Float16)(a1.w * s);
  r[8]  = (_Float16)(b0.x * s); r[9]  = (_Float16)(b0.y * s);
  r[10] = (_Float16)(b0.z * s); r[11] = (_Float16)(b0.w * s);
  r[12] = (_Float16)(b1.x * s); r[13] = (_Float16)(b1.y * s);
  r[14] = (_Float16)(b1.z * s); r[15] = (_Float16)(b1.w * s);
  return r;
}

// ---- 1) per-row inverse L2 norm of x: invn[b*S+s] = 1/max(||x_row||,eps) --
__global__ void k_rownorm(const float* __restrict__ x, float* __restrict__ invn) {
  __shared__ float red[256];
  int row = blockIdx.x, t = threadIdx.x;
  float4 v = *(const float4*)(x + (size_t)row * D_ + t * 4);
  red[t] = v.x * v.x + v.y * v.y + v.z * v.z + v.w * v.w;
  __syncthreads();
  for (int off = 128; off > 0; off >>= 1) {
    if (t < off) red[t] += red[t + off];
    __syncthreads();
  }
  if (t == 0) invn[row] = 1.0f / fmaxf(sqrtf(red[0]), 1e-12f);
}

// ---- 2) xm[b,d] = mean over S (coalesced across d) ------------------------
__global__ void k_colmean(const float* __restrict__ x, float* __restrict__ xm) {
  int gid = blockIdx.x * 256 + threadIdx.x;       // 0 .. B*D-1
  int b = gid >> 10, d = gid & (D_ - 1);
  const float* p = x + (size_t)b * S_ * D_ + d;
  float s = 0.f;
#pragma unroll 4
  for (int t = 0; t < S_; ++t) s += p[(size_t)t * D_];
  xm[gid] = s * (1.0f / (float)S_);
}

// ---- 3) normalized frequencies -> f16 (WMMA B operand for GEMM1) ----------
__global__ void k_fnorm(const float* __restrict__ freq, _Float16* __restrict__ fn16) {
  __shared__ float red[256];
  int r = blockIdx.x, t = threadIdx.x;
  float4 v = *(const float4*)(freq + (size_t)r * D_ + t * 4);
  red[t] = v.x * v.x + v.y * v.y + v.z * v.z + v.w * v.w;
  __syncthreads();
  for (int off = 128; off > 0; off >>= 1) {
    if (t < off) red[t] += red[t + off];
    __syncthreads();
  }
  float inv = 1.0f / fmaxf(sqrtf(red[0]), 1e-12f);
  v4h o;
  o[0] = (_Float16)(v.x * inv); o[1] = (_Float16)(v.y * inv);
  o[2] = (_Float16)(v.z * inv); o[3] = (_Float16)(v.w * inv);
  *(v4h*)(fn16 + (size_t)r * D_ + t * 4) = o;
}

// ---- 4) out_w -> f16 ------------------------------------------------------
__global__ void k_half(const float* __restrict__ src, _Float16* __restrict__ dst) {
  int g = blockIdx.x * 256 + threadIdx.x;
  dst[g] = (_Float16)src[g];
}

// ---- 5) gates & retention: one block per r, coalesced weight-row loads ----
__global__ void k_gates(const float* __restrict__ xm, const float* __restrict__ gw,
                        const float* __restrict__ gb, const float* __restrict__ rw,
                        const float* __restrict__ rb, float* __restrict__ gate,
                        float* __restrict__ ret) {
  __shared__ float red[256];
  int r = blockIdx.x, t = threadIdx.x;
  float4 g4 = *(const float4*)(gw + (size_t)r * D_ + t * 4);
  float4 r4 = *(const float4*)(rw + (size_t)r * D_ + t * 4);
  for (int b = 0; b < B_; ++b) {
    float4 xv = *(const float4*)(xm + (size_t)b * D_ + t * 4);
    float pg = g4.x * xv.x + g4.y * xv.y + g4.z * xv.z + g4.w * xv.w;
    float pr = r4.x * xv.x + r4.y * xv.y + r4.z * xv.z + r4.w * xv.w;
    red[t] = pg; __syncthreads();
    for (int off = 128; off > 0; off >>= 1) { if (t < off) red[t] += red[t + off]; __syncthreads(); }
    float dg = red[0]; __syncthreads();
    red[t] = pr; __syncthreads();
    for (int off = 128; off > 0; off >>= 1) { if (t < off) red[t] += red[t + off]; __syncthreads(); }
    float dr = red[0]; __syncthreads();
    if (t == 0) {
      float gs = 1.f / (1.f + expf(-(dg + gb[r])));
      gate[b * R_ + r] = (gs >= 0.001f) ? gs : 0.f;  // straight-through hard gate (fwd)
      ret[b * R_ + r]  = 1.f / (1.f + expf(-(dr + rb[r])));
    }
  }
}

// ---- 6) GEMM1: signal = (xnorm @ fnorm^T) * gate  [M=B*S, K=D, N=R] -------
// Wave tile 16Mx128N (8 v8f accumulators); block = 8 waves = 64Mx256N.
// K-loop unrolled x2 with explicit ping-pong fragment buffers: each buffer is
// written directly by its prefetch loads and consumed in the alternate phase,
// eliminating the register-rotation copies.
__global__ void __launch_bounds__(256) k_gemm1(const float* __restrict__ x,
                                               const float* __restrict__ invn,
                                               const _Float16* __restrict__ fn16,
                                               const float* __restrict__ gate,
                                               float* __restrict__ signal) {
  int w = threadIdx.x >> 5, lane = threadIdx.x & 31;
  int hl = lane >> 4, lq = lane & 15;
  int wm = w >> 1, wn = w & 1;
  int m_base = blockIdx.x * 64 + wm * 16;     // global row in [0, B*S)
  int n_base = wn * 128;                      // column in [0, R)
  int b = m_base >> 12;                       // row / S (uniform per wave)
  int rowA = m_base + lq;
  const float* ax = x + (size_t)rowA * D_;
  float inv = invn[rowA];
  const _Float16* bbase = fn16 + (size_t)(n_base + lq) * D_;

  v8f c[8] = {};
  const int KSTEPS = D_ / 32;                 // 32 (even)

  auto loadA = [&](int kk) -> v16h {
    return load_frag_f32_scaled(ax + kk * 32 + hl * 8, inv);
  };
  auto loadB = [&](int kk, v16h* bf) {
    int ko = kk * 32 + hl * 8;
#pragma unroll
    for (int n = 0; n < 8; ++n)
      bf[n] = load_frag_h(bbase + (size_t)n * 16 * D_ + ko);
  };

  v16h a0 = loadA(0), a1;
  v16h b0[8], b1[8];
  loadB(0, b0);

  for (int kk = 0; kk < KSTEPS; kk += 2) {
    // phase A: prefetch kk+1, compute kk
    a1 = loadA(kk + 1);
    loadB(kk + 1, b1);
#pragma unroll
    for (int n = 0; n < 8; ++n)
      c[n] = __builtin_amdgcn_wmma_f32_16x16x32_f16(false, a0, false, b0[n],
                                                    (short)0, c[n], false, false);
    // phase B: prefetch kk+2 (tail: redundant in-bounds reload), compute kk+1
    int k2 = (kk + 2 < KSTEPS) ? kk + 2 : kk + 1;
    a0 = loadA(k2);
    loadB(k2, b0);
#pragma unroll
    for (int n = 0; n < 8; ++n)
      c[n] = __builtin_amdgcn_wmma_f32_16x16x32_f16(false, a1, false, b1[n],
                                                    (short)0, c[n], false, false);
  }

#pragma unroll
  for (int n = 0; n < 8; ++n) {
    int r = n_base + n * 16 + lq;
    float g = gate[b * R_ + r];                          // fused gate epilogue
#pragma unroll
    for (int v = 0; v < 8; ++v) {
      int m = m_base + v + 8 * hl;                       // C layout §7.12.2
      signal[(size_t)m * R_ + r] = c[n][v] * g;
    }
  }
}

// ---- 7) linear recurrence over S per (b,r); emits f16 acc + final state ---
__global__ void k_scan(const float* __restrict__ signal, const float* __restrict__ prev,
                       const float* __restrict__ ret, _Float16* __restrict__ acc16,
                       float* __restrict__ fstate) {
  int gid = blockIdx.x * 256 + threadIdx.x;   // 0 .. B*R-1
  int b = gid >> 8, r = gid & (R_ - 1);
  float st = prev[gid];
  float rt = ret[gid];
  const float* sp = signal + (size_t)b * S_ * R_ + r;
  _Float16*    ap = acc16  + (size_t)b * S_ * R_ + r;
#pragma unroll 4
  for (int t = 0; t < S_; ++t) {
    st = fmaf(st, rt, sp[(size_t)t * R_]);
    ap[(size_t)t * R_] = (_Float16)st;
  }
  fstate[gid] = st;
}

// ---- 8) GEMM2: output = acc @ out_w^T  [M=B*S, K=R, N=D] ------------------
// Same ping-pong structure; K loop is 8 steps (even).
__global__ void __launch_bounds__(256) k_gemm2(const _Float16* __restrict__ acc16,
                                               const _Float16* __restrict__ ow16,
                                               float* __restrict__ out) {
  int w = threadIdx.x >> 5, lane = threadIdx.x & 31;
  int hl = lane >> 4, lq = lane & 15;
  int wm = w & 1, wn = w >> 1;
  int m_base = blockIdx.x * 32 + wm * 16;
  int n_base = blockIdx.y * 512 + wn * 128;
  const _Float16* ap = acc16 + (size_t)(m_base + lq) * R_;
  const _Float16* bbase = ow16 + (size_t)(n_base + lq) * R_;

  v8f c[8] = {};
  const int KSTEPS = R_ / 32;                 // 8 (even)

  auto loadA = [&](int kk) -> v16h {
    return load_frag_h(ap + kk * 32 + hl * 8);
  };
  auto loadB = [&](int kk, v16h* bf) {
    int ko = kk * 32 + hl * 8;
#pragma unroll
    for (int n = 0; n < 8; ++n)
      bf[n] = load_frag_h(bbase + (size_t)n * 16 * R_ + ko);
  };

  v16h a0 = loadA(0), a1;
  v16h b0[8], b1[8];
  loadB(0, b0);

  for (int kk = 0; kk < KSTEPS; kk += 2) {
    a1 = loadA(kk + 1);
    loadB(kk + 1, b1);
#pragma unroll
    for (int n = 0; n < 8; ++n)
      c[n] = __builtin_amdgcn_wmma_f32_16x16x32_f16(false, a0, false, b0[n],
                                                    (short)0, c[n], false, false);
    int k2 = (kk + 2 < KSTEPS) ? kk + 2 : kk + 1;
    a0 = loadA(k2);
    loadB(k2, b0);
#pragma unroll
    for (int n = 0; n < 8; ++n)
      c[n] = __builtin_amdgcn_wmma_f32_16x16x32_f16(false, a1, false, b1[n],
                                                    (short)0, c[n], false, false);
  }

#pragma unroll
  for (int n = 0; n < 8; ++n) {
    int d = n_base + n * 16 + lq;
#pragma unroll
    for (int v = 0; v < 8; ++v) {
      int m = m_base + v + 8 * hl;
      out[(size_t)m * D_ + d] = c[n][v];
    }
  }
}

extern "C" void kernel_launch(void* const* d_in, const int* in_sizes, int n_in,
                              void* d_out, int out_size, void* d_ws, size_t ws_size,
                              hipStream_t stream) {
  (void)in_sizes; (void)n_in; (void)out_size; (void)ws_size;
  const float* x    = (const float*)d_in[0];   // (B,S,D)
  const float* prev = (const float*)d_in[1];   // (B,R)
  const float* freq = (const float*)d_in[2];   // (R,D)
  const float* rb   = (const float*)d_in[3];   // (R,)
  const float* rw   = (const float*)d_in[4];   // (R,D)
  const float* gw   = (const float*)d_in[5];   // (R,D)
  const float* gb   = (const float*)d_in[6];   // (R,)
  const float* ow   = (const float*)d_in[7];   // (D,R)

  float* out    = (float*)d_out;                       // (B,S,D) flat
  float* fstate = out + (size_t)B_ * S_ * D_;          // (B,R) appended

  // Workspace carve (~52 MB), 256B-aligned slices.
  char* p = (char*)d_ws;
  auto carve = [&](size_t bytes) -> void* {
    void* r = (void*)p;
    p += (bytes + 255) & ~(size_t)255;
    return r;
  };
  float*    invn   = (float*)   carve((size_t)B_ * S_ * 4);
  float*    xm     = (float*)   carve((size_t)B_ * D_ * 4);
  _Float16* fn16   = (_Float16*)carve((size_t)R_ * D_ * 2);
  _Float16* ow16   = (_Float16*)carve((size_t)D_ * R_ * 2);
  float*    gate   = (float*)   carve((size_t)B_ * R_ * 4);
  float*    ret    = (float*)   carve((size_t)B_ * R_ * 4);
  float*    signal = (float*)   carve((size_t)B_ * S_ * R_ * 4);
  _Float16* acc16  = (_Float16*)carve((size_t)B_ * S_ * R_ * 2);

  k_rownorm<<<B_ * S_, 256, 0, stream>>>(x, invn);
  k_colmean<<<(B_ * D_) / 256, 256, 0, stream>>>(x, xm);
  k_fnorm  <<<R_, 256, 0, stream>>>(freq, fn16);
  k_half   <<<(D_ * R_) / 256, 256, 0, stream>>>(ow, ow16);
  k_gates  <<<R_, 256, 0, stream>>>(xm, gw, gb, rw, rb, gate, ret);
  k_gemm1  <<<(B_ * S_) / 64, 256, 0, stream>>>(x, invn, fn16, gate, signal);
  k_scan   <<<(B_ * R_) / 256, 256, 0, stream>>>(signal, prev, ret, acc16, fstate);
  k_gemm2  <<<dim3((B_ * S_) / 32, D_ / 512), 256, 0, stream>>>(acc16, ow16, out);
}
